// NovelReg_31430570672575
// MI455X (gfx1250) — matrix operand order repaired and checked by reading, not compiled, not run on hardware
//
#include <hip/hip_runtime.h>

#define NN  50000
#define EE  800000
#define NG  64
#define E2  (EE + NN)

typedef __attribute__((ext_vector_type(16))) _Float16 v16h;
typedef __attribute__((ext_vector_type(8)))  _Float16 v8h;
typedef __attribute__((ext_vector_type(8)))  float    v8f;

// ---- float <-> order-preserving unsigned (for atomic max; 0 == "-inf") ----
__device__ __forceinline__ unsigned ford(float f) {
  unsigned u = __float_as_uint(f);
  return (u & 0x80000000u) ? ~u : (u | 0x80000000u);
}
__device__ __forceinline__ float funord(unsigned u) {
  unsigned b = (u & 0x80000000u) ? (u & 0x7fffffffu) : ~u;
  return __uint_as_float(b);
}
__device__ __forceinline__ float wave_sum(float v) {
#pragma unroll
  for (int off = 16; off; off >>= 1) v += __shfl_xor(v, off, 32);
  return v;
}
__device__ __forceinline__ void f4_to_arr(float4 f, float* o) {
  o[0] = f.x; o[1] = f.y; o[2] = f.z; o[3] = f.w;
}
__device__ __forceinline__ float4 arr_to_f4(const float* o) {
  return make_float4(o[0], o[1], o[2], o[3]);
}

// ================= WMMA GEMM =================
// C[nrows x Nout] = X[nrows x K] @ W[K x Nout] + bias.
// Block = 256 threads (8 waves). Block covers 128 rows x 64 cols.
// The 64-col weight panel is staged once per block into LDS, transposed and
// pre-converted to f16: Wt[c][k], row-padded to K+8 halves. With the CDNA5
// B-fragment layout (lane holds K 8h..8h+7 and 16+8h..16+8h+7 at fixed col), each B
// fragment is two 16B LDS reads 32B apart (2x ds_load_b128), bank-spread by
// the padding. Each wave computes a 16x64 strip: one A fragment (float2
// global loads) feeds 4 WMMAs per 32-K step; K templated so the loop fully
// unrolls.
template <int K>
__global__ void wmma_gemm_t(const float* __restrict__ X, const float* __restrict__ W,
                            const float* __restrict__ bias, float* __restrict__ C,
                            int nrows, int Nout) {
  constexpr int KP = K + 8;
  extern __shared__ _Float16 smem[];  // Wt[64][KP]
  int n0 = blockIdx.y * 64;
  for (int i = threadIdx.x; i < 64 * K; i += 256) {
    int k = i >> 6, c = i & 63;  // consecutive threads -> consecutive cols (coalesced)
    smem[c * KP + k] = (_Float16)W[(size_t)k * Nout + n0 + c];
  }
  __syncthreads();

  int wv   = threadIdx.x >> 5;
  int lane = threadIdx.x & 31;
  int m0   = (blockIdx.x * 8 + wv) * 16;
  if (m0 >= nrows) return;
  int lm = lane & 15;
  int hi = lane >> 4;

  const float2* Xr = (const float2*)(X + (size_t)(m0 + lm) * K);
  v8f acc0 = {}, acc1 = {}, acc2 = {}, acc3 = {};
#pragma unroll
  for (int kb = 0; kb < K; kb += 32) {
    // A fragment (16x32 f16, ISA 7.12.2): lane M = lane&15;
    // VGPR v holds K = kb + (v&3)*2 + hi*8 + (v>=4)*16 (consecutive pairs -> b64 loads)
    v16h a;
#pragma unroll
    for (int v = 0; v < 8; ++v) {
      int kA = kb + ((v & 3) << 1) + (hi << 3) + ((v >> 2) << 4);
      float2 t = Xr[kA >> 1];
      a[2 * v]     = (_Float16)t.x;
      a[2 * v + 1] = (_Float16)t.y;
    }
    const _Float16* brow = smem + kb + (hi << 3);
#pragma unroll
    for (int t = 0; t < 4; ++t) {
      const _Float16* bp = brow + (t * 16 + lm) * KP;  // 16B-aligned
      v8h blo = *(const v8h*)bp;
      v8h bhi = *(const v8h*)(bp + 16);
      v16h b;
#pragma unroll
      for (int j = 0; j < 8; ++j) { b[j] = blo[j]; b[8 + j] = bhi[j]; }
      v8f& acc = (t == 0) ? acc0 : (t == 1) ? acc1 : (t == 2) ? acc2 : acc3;
      acc = __builtin_amdgcn_wmma_f32_16x16x32_f16(false, a, false, b, (short)0, acc,
                                                   false, false);
    }
  }
  // C/D layout: lanes 0-15 rows m0+r, lanes 16-31 rows m0+8+r, col = tile*16 + lm
#pragma unroll
  for (int t = 0; t < 4; ++t) {
    v8f acc = (t == 0) ? acc0 : (t == 1) ? acc1 : (t == 2) ? acc2 : acc3;
    int col = n0 + t * 16 + lm;
    float bv = bias ? bias[col] : 0.0f;
#pragma unroll
    for (int r = 0; r < 8; ++r) {
      int row = m0 + (hi << 3) + r;
      C[(size_t)row * Nout + col] = acc[r] + bv;
    }
  }
}

// ================= GATv2 edge kernels (wave per edge; self-loops appended logically) =================
__global__ void gat_edge_logits(const float* __restrict__ xl, const float* __restrict__ xr,
                                const float* __restrict__ att, const int* __restrict__ src,
                                const int* __restrict__ dst, float* __restrict__ wbuf,
                                unsigned* __restrict__ nodemax) {
  int wid  = (blockIdx.x * blockDim.x + threadIdx.x) >> 5;
  int lane = threadIdx.x & 31;
  if (wid >= E2) return;
  int s, d;
  if (wid < EE) { s = src[wid]; d = dst[wid]; } else { s = wid - EE; d = s; }
  // hide HBM latency on the gather rows of edges ~1024 ahead
  int pf = wid + 1024;
  if (pf < E2) {
    int ps, pd;
    if (pf < EE) { ps = src[pf]; pd = dst[pf]; } else { ps = pd = pf - EE; }
    if (lane < 8) {
      __builtin_prefetch(xl + (size_t)ps * 256 + lane * 32, 0, 1);
      __builtin_prefetch(xr + (size_t)pd * 256 + lane * 32, 0, 1);
    }
  }
  const float4* a = (const float4*)(xl + (size_t)s * 256);
  const float4* r = (const float4*)(xr + (size_t)d * 256);
  const float4* A = (const float4*)att;
  float p0 = 0.f, p1 = 0.f;
  {
    float av[4], rv[4], tv[4];
    f4_to_arr(a[lane], av); f4_to_arr(r[lane], rv); f4_to_arr(A[lane], tv);
#pragma unroll
    for (int j = 0; j < 4; ++j) {
      float e = av[j] + rv[j];
      e = e > 0.f ? e : 0.2f * e;
      p0 += e * tv[j];
    }
    f4_to_arr(a[lane + 32], av); f4_to_arr(r[lane + 32], rv); f4_to_arr(A[lane + 32], tv);
#pragma unroll
    for (int j = 0; j < 4; ++j) {
      float e = av[j] + rv[j];
      e = e > 0.f ? e : 0.2f * e;
      p1 += e * tv[j];
    }
  }
  p0 = wave_sum(p0);
  p1 = wave_sum(p1);
  if (lane == 0) {
    ((float2*)wbuf)[wid] = make_float2(p0, p1);
    atomicMax(&nodemax[d * 2],     ford(p0));
    atomicMax(&nodemax[d * 2 + 1], ford(p1));
  }
}

__global__ void gat_edge_expw(const int* __restrict__ dst, const unsigned* __restrict__ nodemax,
                              float* __restrict__ wbuf, float* __restrict__ nodesum) {
  int e = blockIdx.x * blockDim.x + threadIdx.x;
  if (e >= E2) return;
  int d = (e < EE) ? dst[e] : (e - EE);
  unsigned k0 = nodemax[d * 2], k1 = nodemax[d * 2 + 1];
  float m0 = (k0 == 0u) ? 0.f : funord(k0);
  float m1 = (k1 == 0u) ? 0.f : funord(k1);
  float2 lw = ((float2*)wbuf)[e];
  lw.x = __expf(lw.x - m0);
  lw.y = __expf(lw.y - m1);
  ((float2*)wbuf)[e] = lw;
  atomicAdd(&nodesum[d * 2],     lw.x);
  atomicAdd(&nodesum[d * 2 + 1], lw.y);
}

__global__ void gat_edge_scatter(const float* __restrict__ xl, const int* __restrict__ src,
                                 const int* __restrict__ dst, const float* __restrict__ wbuf,
                                 const float* __restrict__ nodesum, float* __restrict__ acc) {
  int wid  = (blockIdx.x * blockDim.x + threadIdx.x) >> 5;
  int lane = threadIdx.x & 31;
  if (wid >= E2) return;
  int s, d;
  if (wid < EE) { s = src[wid]; d = dst[wid]; } else { s = wid - EE; d = s; }
  int pf = wid + 1024;
  if (pf < E2) {
    int ps = (pf < EE) ? src[pf] : (pf - EE);
    if (lane < 8) __builtin_prefetch(xl + (size_t)ps * 256 + lane * 32, 0, 1);
  }
  float2 lw = ((const float2*)wbuf)[wid];
  float a0 = lw.x / fmaxf(nodesum[d * 2],     1e-16f);
  float a1 = lw.y / fmaxf(nodesum[d * 2 + 1], 1e-16f);
  const float4* xs = (const float4*)(xl + (size_t)s * 256);
  float x0[4], x1[4];
  f4_to_arr(xs[lane], x0);
  f4_to_arr(xs[lane + 32], x1);
  float* out = acc + (size_t)d * 256 + lane * 4;
#pragma unroll
  for (int j = 0; j < 4; ++j) {
    atomicAdd(&out[j],       a0 * x0[j]);
    atomicAdd(&out[j + 128], a1 * x1[j]);
  }
}

// mean over heads + bias, ReLU, optional LayerNorm -> hout[N x 128]; wave per node
__global__ void gat_node_finish(const float* __restrict__ acc, const float* __restrict__ bias,
                                const float* __restrict__ ln_g, const float* __restrict__ ln_b,
                                int do_ln, float* __restrict__ hout) {
  int n    = (blockIdx.x * blockDim.x + threadIdx.x) >> 5;
  int lane = threadIdx.x & 31;
  if (n >= NN) return;
  const float4* an = (const float4*)(acc + (size_t)n * 256);
  float h0[4], h1[4], bv[4], v[4];
  f4_to_arr(an[lane], h0);
  f4_to_arr(an[lane + 32], h1);
  f4_to_arr(((const float4*)bias)[lane], bv);
#pragma unroll
  for (int j = 0; j < 4; ++j) {
    float t = 0.5f * (h0[j] + h1[j]) + bv[j];
    v[j] = t > 0.f ? t : 0.f;
  }
  if (do_ln) {
    float s = wave_sum(v[0] + v[1] + v[2] + v[3]);
    float mu = s * (1.0f / 128.0f);
    float q = 0.f;
#pragma unroll
    for (int j = 0; j < 4; ++j) { float dd = v[j] - mu; q += dd * dd; }
    q = wave_sum(q);
    float inv = rsqrtf(q * (1.0f / 128.0f) + 1e-5f);
    float g[4], b[4];
    f4_to_arr(((const float4*)ln_g)[lane], g);
    f4_to_arr(((const float4*)ln_b)[lane], b);
#pragma unroll
    for (int j = 0; j < 4; ++j) v[j] = (v[j] - mu) * inv * g[j] + b[j];
  }
  ((float4*)(hout + (size_t)n * 128))[lane] = arr_to_f4(v);
}

// ================= FiLM =================
__global__ void film_edge(const float* __restrict__ hw, const float* __restrict__ bg,
                          const int* __restrict__ src, const int* __restrict__ dst,
                          float* __restrict__ agg, int* __restrict__ cnt) {
  int wid  = (blockIdx.x * blockDim.x + threadIdx.x) >> 5;
  int lane = threadIdx.x & 31;
  if (wid >= EE) return;
  int s = src[wid], d = dst[wid];
  int pf = wid + 1024;
  if (pf < EE) {
    int ps = src[pf], pd = dst[pf];
    if (lane < 4) __builtin_prefetch(hw + (size_t)ps * 128 + lane * 32, 0, 1);
    if (lane < 8) __builtin_prefetch(bg + (size_t)pd * 256 + lane * 32, 0, 1);
  }
  float hs[4], be[4], ga[4];
  f4_to_arr(((const float4*)(hw + (size_t)s * 128))[lane], hs);
  const float4* bgd = (const float4*)(bg + (size_t)d * 256);  // beta [0,128), gamma [128,256)
  f4_to_arr(bgd[lane], be);
  f4_to_arr(bgd[lane + 32], ga);
  float* ag = agg + (size_t)d * 128 + lane * 4;
#pragma unroll
  for (int j = 0; j < 4; ++j) {
    float m = ga[j] * hs[j] + be[j];
    atomicAdd(&ag[j], m > 0.f ? m : 0.f);
  }
  if (lane == 0) atomicAdd(&cnt[d], 1);
}

__global__ void film_node(const float* __restrict__ bgs, const float* __restrict__ sx,
                          const float* __restrict__ agg, const int* __restrict__ cnt,
                          const float* __restrict__ ln_g, const float* __restrict__ ln_b,
                          float* __restrict__ hout) {
  int n    = (blockIdx.x * blockDim.x + threadIdx.x) >> 5;
  int lane = threadIdx.x & 31;
  if (n >= NN) return;
  const float4* bn = (const float4*)(bgs + (size_t)n * 256);
  float be[4], ga[4], sn[4], an[4], v[4];
  f4_to_arr(bn[lane], be);
  f4_to_arr(bn[lane + 32], ga);
  f4_to_arr(((const float4*)(sx + (size_t)n * 128))[lane], sn);
  f4_to_arr(((const float4*)(agg + (size_t)n * 128))[lane], an);
  float invc = 1.0f / fmaxf((float)cnt[n], 1.0f);
#pragma unroll
  for (int j = 0; j < 4; ++j) {
    float self = ga[j] * sn[j] + be[j];
    self = self > 0.f ? self : 0.f;
    float t = self + an[j] * invc;
    v[j] = t > 0.f ? t : 0.f;  // relu before LN
  }
  float s = wave_sum(v[0] + v[1] + v[2] + v[3]);
  float mu = s * (1.0f / 128.0f);
  float q = 0.f;
#pragma unroll
  for (int j = 0; j < 4; ++j) { float dd = v[j] - mu; q += dd * dd; }
  q = wave_sum(q);
  float inv = rsqrtf(q * (1.0f / 128.0f) + 1e-5f);
  float g[4], b[4];
  f4_to_arr(((const float4*)ln_g)[lane], g);
  f4_to_arr(((const float4*)ln_b)[lane], b);
#pragma unroll
  for (int j = 0; j < 4; ++j) v[j] = (v[j] - mu) * inv * g[j] + b[j];
  ((float4*)(hout + (size_t)n * 128))[lane] = arr_to_f4(v);
}

// ================= Pool + final projection =================
__global__ void pool_scatter(const float* __restrict__ h, const int* __restrict__ batch,
                             float* __restrict__ pooled, int* __restrict__ gcnt) {
  int n    = (blockIdx.x * blockDim.x + threadIdx.x) >> 5;
  int lane = threadIdx.x & 31;
  if (n >= NN) return;
  int g = batch[n];
  float hv[4];
  f4_to_arr(((const float4*)(h + (size_t)n * 128))[lane], hv);
  float* p = pooled + (size_t)g * 128 + lane * 4;
#pragma unroll
  for (int j = 0; j < 4; ++j) atomicAdd(&p[j], hv[j]);
  if (lane == 0) atomicAdd(&gcnt[g], 1);
}

__global__ void pool_final(const float* __restrict__ pooled, const int* __restrict__ gcnt,
                           const float* __restrict__ mpW, const float* __restrict__ mpb,
                           float* __restrict__ out) {
  int g    = (blockIdx.x * blockDim.x + threadIdx.x) >> 5;
  int lane = threadIdx.x & 31;
  if (g >= NG) return;
  float invc = 1.0f / fmaxf((float)gcnt[g], 1.0f);
  float p[4], w[4];
  f4_to_arr(((const float4*)(pooled + (size_t)g * 128))[lane], p);
  f4_to_arr(((const float4*)mpW)[lane], w);
  float s = (p[0] * w[0] + p[1] * w[1] + p[2] * w[2] + p[3] * w[3]) * invc;
  s = wave_sum(s);
  if (lane == 0) out[g] = s + mpb[0];
}

// ================= Host orchestration =================
extern "C" void kernel_launch(void* const* d_in, const int* in_sizes, int n_in,
                              void* d_out, int out_size, void* d_ws, size_t ws_size,
                              hipStream_t stream) {
  (void)in_sizes; (void)n_in; (void)out_size; (void)ws_size;
  const float* x      = (const float*)d_in[0];
  const int*   edge   = (const int*)d_in[1];
  const int*   batch  = (const int*)d_in[2];
  const float* g0_Wl  = (const float*)d_in[3];
  const float* g0_bl  = (const float*)d_in[4];
  const float* g0_Wr  = (const float*)d_in[5];
  const float* g0_br  = (const float*)d_in[6];
  const float* g0_att = (const float*)d_in[7];
  const float* g0_b   = (const float*)d_in[8];
  const float* g2_Wl  = (const float*)d_in[9];
  const float* g2_bl  = (const float*)d_in[10];
  const float* g2_Wr  = (const float*)d_in[11];
  const float* g2_br  = (const float*)d_in[12];
  const float* g2_att = (const float*)d_in[13];
  const float* g2_b   = (const float*)d_in[14];
  const float* g3_Wl  = (const float*)d_in[15];
  const float* g3_bl  = (const float*)d_in[16];
  const float* g3_Wr  = (const float*)d_in[17];
  const float* g3_br  = (const float*)d_in[18];
  const float* g3_att = (const float*)d_in[19];
  const float* g3_b   = (const float*)d_in[20];
  const float* f1_W   = (const float*)d_in[21];
  const float* f1_Ws  = (const float*)d_in[22];
  const float* f1_Wf  = (const float*)d_in[23];
  const float* f1_bf  = (const float*)d_in[24];
  const float* f1_Wfs = (const float*)d_in[25];
  const float* f1_bfs = (const float*)d_in[26];
  const float* ln0_g  = (const float*)d_in[27];
  const float* ln0_b  = (const float*)d_in[28];
  const float* ln1_g  = (const float*)d_in[29];
  const float* ln1_b  = (const float*)d_in[30];
  const float* ln2_g  = (const float*)d_in[31];
  const float* ln2_b  = (const float*)d_in[32];
  const float* mp_W   = (const float*)d_in[33];
  const float* mp_b   = (const float*)d_in[34];

  const int* src = edge;
  const int* dst = edge + EE;

  // workspace carve-out (all 4-byte elements)
  float* base = (float*)d_ws;
  size_t o = 0;
  float*    xl      = base + o; o += (size_t)NN * 256;
  float*    xr      = base + o; o += (size_t)NN * 256;
  float*    acc     = base + o; o += (size_t)NN * 256;
  float*    h0      = base + o; o += (size_t)NN * 128;
  float*    h1      = base + o; o += (size_t)NN * 128;
  float*    agg     = base + o; o += (size_t)NN * 128;
  float*    wbuf    = base + o; o += (size_t)E2 * 2;
  unsigned* nodemax = (unsigned*)(base + o); o += (size_t)NN * 2;
  float*    nodesum = base + o; o += (size_t)NN * 2;
  int*      cnt     = (int*)(base + o); o += NN;
  float*    pooled  = base + o; o += (size_t)NG * 128;
  int*      gcnt    = (int*)(base + o); o += NG;

  dim3 blk(256);
  auto wgrid = [](long nwaves) { return dim3((unsigned)((nwaves + 7) / 8)); };
  const unsigned gemmBlksM = (3125 + 7) / 8;  // tilesM = NN/16 = 3125

  auto launch_gemm = [&](const float* Xp, const float* Wp, const float* bp, float* Cp,
                         int K, int Nout) {
    dim3 g(gemmBlksM, (unsigned)(Nout / 64));
    size_t sh = (size_t)64 * (K + 8) * sizeof(_Float16);
    if (K == 64)
      wmma_gemm_t<64><<<g, blk, sh, stream>>>(Xp, Wp, bp, Cp, NN, Nout);
    else
      wmma_gemm_t<128><<<g, blk, sh, stream>>>(Xp, Wp, bp, Cp, NN, Nout);
  };

  auto run_gat = [&](const float* hin, int K, const float* Wl, const float* bl,
                     const float* Wr, const float* br, const float* att,
                     const float* bias, const float* lng, const float* lnb,
                     int do_ln, float* hout) {
    launch_gemm(hin, Wl, bl, xl, K, 256);
    launch_gemm(hin, Wr, br, xr, K, 256);
    hipMemsetAsync(nodemax, 0, (size_t)NN * 2 * 4, stream);
    hipMemsetAsync(nodesum, 0, (size_t)NN * 2 * 4, stream);
    hipMemsetAsync(acc, 0, (size_t)NN * 256 * 4, stream);
    gat_edge_logits<<<wgrid(E2), blk, 0, stream>>>(xl, xr, att, src, dst, wbuf, nodemax);
    gat_edge_expw<<<dim3((E2 + 255) / 256), blk, 0, stream>>>(dst, nodemax, wbuf, nodesum);
    gat_edge_scatter<<<wgrid(E2), blk, 0, stream>>>(xl, src, dst, wbuf, nodesum, acc);
    gat_node_finish<<<wgrid(NN), blk, 0, stream>>>(acc, bias, lng, lnb, do_ln, hout);
  };

  // layer 0: GATv2(x) -> relu -> LN0 -> h0
  run_gat(x, 64, g0_Wl, g0_bl, g0_Wr, g0_br, g0_att, g0_b, ln0_g, ln0_b, 1, h0);

  // layer 1: FiLM(h0) -> relu -> LN1 -> h1   (reuse xl/xr/acc as bgs/bg/{s,hw})
  {
    float* bgs = xl;
    float* bg  = xr;
    float* sx  = acc;
    float* hw  = acc + (size_t)NN * 128;
    launch_gemm(h0, f1_Wfs, f1_bfs, bgs, 128, 256);
    launch_gemm(h0, f1_Ws, nullptr, sx, 128, 128);
    launch_gemm(h0, f1_Wf, f1_bf, bg, 128, 256);
    launch_gemm(h0, f1_W, nullptr, hw, 128, 128);
    hipMemsetAsync(agg, 0, (size_t)NN * 128 * 4, stream);
    hipMemsetAsync(cnt, 0, (size_t)NN * 4, stream);
    film_edge<<<wgrid(EE), blk, 0, stream>>>(hw, bg, src, dst, agg, cnt);
    film_node<<<wgrid(NN), blk, 0, stream>>>(bgs, sx, agg, cnt, ln1_g, ln1_b, h1);
  }

  // layer 2: GATv2(h1) -> relu -> LN2 -> h0
  run_gat(h1, 128, g2_Wl, g2_bl, g2_Wr, g2_br, g2_att, g2_b, ln2_g, ln2_b, 1, h0);

  // layer 3: GATv2(h0) -> relu -> h1 (no LN)
  run_gat(h0, 128, g3_Wl, g3_bl, g3_Wr, g3_br, g3_att, g3_b, ln0_g, ln0_b, 0, h1);

  // mean pool per graph, then 128->1 projection
  hipMemsetAsync(pooled, 0, (size_t)NG * 128 * 4, stream);
  hipMemsetAsync(gcnt, 0, (size_t)NG * 4, stream);
  pool_scatter<<<wgrid(NN), blk, 0, stream>>>(h1, batch, pooled, gcnt);
  pool_final<<<dim3(8), blk, 0, stream>>>(pooled, gcnt, mp_W, mp_b, (float*)d_out);
}
